// GINBaseline_32538672234672
// MI455X (gfx1250) — matrix operand order repaired and checked by reading, not compiled, not run on hardware
//
#include <hip/hip_runtime.h>
#include <stdint.h>

// ---------------- problem constants (match reference) ----------------
#define N_NODES   50000
#define N_EDGES   800000
#define DIM       128
#define DIM2      256
#define NCLS      10
#define NGRAPH    64
#define NLAYER    3
#define BN_EPS    1e-5f

typedef __attribute__((ext_vector_type(16))) __bf16 v16bf;
typedef __attribute__((ext_vector_type(8)))  float  v8f;

union FragBF { v16bf v; uint32_t u[8]; };

// fp32 -> bf16 bits, round-to-nearest-even
__device__ __forceinline__ uint32_t f2bf(float f) {
  uint32_t u = __float_as_uint(f);
  return (u + 0x7FFFu + ((u >> 16) & 1u)) >> 16;
}

// ---------------- BN folding: scale = g*rsqrt(v+eps), shift = (b-m)*scale+bt --
__global__ void prep_params(const float* __restrict__ be,
                            const float* __restrict__ b1, const float* __restrict__ g1,
                            const float* __restrict__ bt1, const float* __restrict__ m1,
                            const float* __restrict__ v1,
                            const float* __restrict__ b2, const float* __restrict__ g2,
                            const float* __restrict__ bt2, const float* __restrict__ m2,
                            const float* __restrict__ v2,
                            float* __restrict__ embS, float* __restrict__ embH,
                            float* __restrict__ s1, float* __restrict__ h1,
                            float* __restrict__ s2, float* __restrict__ h2) {
  int j = threadIdx.x;                    // 256 threads, one block
  if (j < DIM) { embS[j] = 1.0f; embH[j] = be[j]; }
  for (int l = 0; l < NLAYER; ++l) {
    { int i = l * DIM2 + j;
      float s = g1[i] * rsqrtf(v1[i] + BN_EPS);
      s1[i] = s; h1[i] = (b1[i] - m1[i]) * s + bt1[i]; }
    if (j < DIM) { int i = l * DIM + j;
      float s = g2[i] * rsqrtf(v2[i] + BN_EPS);
      s2[i] = s; h2[i] = (b2[i] - m2[i]) * s + bt2[i]; }
  }
}

// ---------------- pack fp32 W[K][Nout] into WMMA B-fragment bf16 layout ------
// lane holds B[kbase+2v..2v+1][nt*16 + lane%16], kbase = kt*32 + 16*(lane/16)
__global__ void pack_w(const float* __restrict__ W, uint32_t* __restrict__ out,
                       int K, int Nout) {
  const int Ktiles = K >> 5;
  const int kt = blockIdx.x % Ktiles;
  const int nt = blockIdx.x / Ktiles;
  const int lane = threadIdx.x;           // 32 threads
  const int n = (nt << 4) + (lane & 15);
  const int kbase = (kt << 5) + ((lane >> 4) << 4);
  uint32_t* o = out + (((size_t)nt * Ktiles + kt) * 32 + lane) * 8;
#pragma unroll
  for (int j = 0; j < 8; ++j) {
    const int k0 = kbase + 2 * j;
    const uint32_t lo = f2bf(W[(size_t)k0 * Nout + n]);
    const uint32_t hi = f2bf(W[(size_t)(k0 + 1) * Nout + n]);
    o[j] = lo | (hi << 16);
  }
}

// ---------------- fp32 -> packed bf16 pairs ----------------------------------
__global__ void f32_to_bf16(const float2* __restrict__ in, uint32_t* __restrict__ out,
                            int ndw) {
  int i = blockIdx.x * 256 + threadIdx.x;
  if (i < ndw) { float2 f = in[i]; out[i] = f2bf(f.x) | (f2bf(f.y) << 16); }
}

// ---------------- z = h (copy, float4) ---------------------------------------
__global__ void copy_f4(const float4* __restrict__ in, float4* __restrict__ out, int n4) {
  int i = blockIdx.x * 256 + threadIdx.x;
  if (i < n4) out[i] = in[i];
}

__global__ void zero_buf(float* __restrict__ p, int n) {
  int i = blockIdx.x * 256 + threadIdx.x;
  if (i < n) p[i] = 0.0f;
}

// ---------------- edge scatter: z[u] += h[c]  (L2-resident atomics) ----------
__global__ void scatter_add(const float* __restrict__ h, const int* __restrict__ ce,
                            const int* __restrict__ ue, float* __restrict__ z) {
  int tid = blockIdx.x * 256 + threadIdx.x;        // E*32 threads, float4 each
  if (tid >= N_EDGES * 32) return;
  const int e = tid >> 5;
  const int q = (tid & 31) << 2;
  const int s = ce[e], d = ue[e];
  const float4 v = *(const float4*)(h + (size_t)s * DIM + q);
  float* p = z + (size_t)d * DIM + q;
  atomicAdd(p + 0, v.x); atomicAdd(p + 1, v.y);
  atomicAdd(p + 2, v.z); atomicAdd(p + 3, v.w);
}

// ---------------- graph pooling: pooled[batch[i]] += h[i] --------------------
__global__ void pool_add(const float* __restrict__ h, const int* __restrict__ batch,
                         float* __restrict__ pooled) {
  int tid = blockIdx.x * 256 + threadIdx.x;        // N*32 threads
  if (tid >= N_NODES * 32) return;
  const int i = tid >> 5;
  const int q = (tid & 31) << 2;
  const int g = batch[i];
  const float4 v = *(const float4*)(h + (size_t)i * DIM + q);
  float* p = pooled + (size_t)g * DIM + q;
  atomicAdd(p + 0, v.x); atomicAdd(p + 1, v.y);
  atomicAdd(p + 2, v.z); atomicAdd(p + 3, v.w);
}

// ---------------- WMMA GEMM: out = act(A_bf16 @ Wpacked * scale + shift) -----
// block = 256 threads (8 waves), owns 16 rows; wave owns 16-col tiles.
// A staged in LDS (bf16 pairs, padded rows), B read as packed fragments.
template <bool RELU, bool OUT_BF16>
__global__ __launch_bounds__(256) void gemm_wmma(
    const uint32_t* __restrict__ A,     // bf16 pairs, M x K row-major
    const uint32_t* __restrict__ Bp,    // packed fragments
    const float* __restrict__ scale, const float* __restrict__ shift,
    void* __restrict__ outv, int K, int Nout) {
  __shared__ uint32_t ldsA[16 * 132];              // up to K=256 (128 dw) + pad 4
  const int tid = threadIdx.x;
  const int wave = tid >> 5, lane = tid & 31;
  const int m = lane & 15, half = lane >> 4;
  const int rowStart = blockIdx.x << 4;
  const int Kd = K >> 1;                           // dwords per A row
  const int rs = Kd + 4;                           // padded LDS row stride
  for (int e = tid; e < 16 * Kd; e += 256) {
    const int r = e / Kd, c = e - r * Kd;
    ldsA[r * rs + c] = A[(size_t)(rowStart + r) * Kd + c];
  }
  __syncthreads();
  const int Ktiles = K >> 5;
  const int Ntiles = Nout >> 4;
  for (int nt = wave; nt < Ntiles; nt += 8) {
    v8f acc = {};
    const uint32_t* bp = Bp + (((size_t)nt * Ktiles) * 32 + lane) * 8;
    for (int kt = 0; kt < Ktiles; ++kt) {
      FragBF a, b;
      // A 16x32 layout (matches packed B): lanes0-15 K 0-15, lanes16-31 K 16-31
      const int base = m * rs + (kt << 4) + (half << 3);
#pragma unroll
      for (int i = 0; i < 4; ++i) {
        a.u[i]     = ldsA[base + i];
        a.u[4 + i] = ldsA[base + 4 + i];
      }
      const uint32_t* bt = bp + (size_t)kt * 256;
#pragma unroll
      for (int i = 0; i < 8; ++i) b.u[i] = bt[i];
      acc = __builtin_amdgcn_wmma_f32_16x16x32_bf16(
          false, a.v, false, b.v, (short)0, acc, false, false);
    }
    const int col = (nt << 4) + m;
    const float sc = scale[col], sh = shift[col];
#pragma unroll
    for (int v = 0; v < 8; ++v) {                  // row = v + 8*half (ISA C/D layout)
      float o = acc[v] * sc + sh;
      if (RELU) o = fmaxf(o, 0.0f);
      const size_t idx = (size_t)(rowStart + (half << 3) + v) * Nout + col;
      if (OUT_BF16) ((uint16_t*)outv)[idx] = (uint16_t)f2bf(o);
      else          ((float*)outv)[idx] = o;
    }
  }
}

// ---------------- tiny readout: out = relu(pooled@Wr1+br1)@Wr2+br2 -----------
__global__ void readout(const float* __restrict__ pooled,
                        const float* __restrict__ Wr1, const float* __restrict__ br1,
                        const float* __restrict__ Wr2, const float* __restrict__ br2,
                        float* __restrict__ out) {
  __shared__ float p[DIM], t[DIM];
  const int g = blockIdx.x, j = threadIdx.x;       // 64 blocks x 128 threads
  p[j] = pooled[(size_t)g * DIM + j];
  __syncthreads();
  float acc = br1[j];
  for (int k = 0; k < DIM; ++k) acc += p[k] * Wr1[(size_t)k * DIM + j];
  t[j] = fmaxf(acc, 0.0f);
  __syncthreads();
  if (j < NCLS) {
    float o = br2[j];
    for (int k = 0; k < DIM; ++k) o += t[k] * Wr2[(size_t)k * NCLS + j];
    out[(size_t)g * NCLS + j] = o;
  }
}

// ---------------- host side ---------------------------------------------------
static inline size_t alignup(size_t x) { return (x + 255) & ~(size_t)255; }

extern "C" void kernel_launch(void* const* d_in, const int* in_sizes, int n_in,
                              void* d_out, int out_size, void* d_ws, size_t ws_size,
                              hipStream_t stream) {
  (void)in_sizes; (void)n_in; (void)out_size; (void)ws_size;
  // inputs in setup_inputs() order
  const float* x    = (const float*)d_in[0];
  const int*   c2   = (const int*)  d_in[1];
  const int*   u2   = (const int*)  d_in[2];
  const int*   batch= (const int*)  d_in[3];
  const float* We   = (const float*)d_in[4];
  const float* be   = (const float*)d_in[5];
  const float* W1   = (const float*)d_in[6];
  const float* b1   = (const float*)d_in[7];
  const float* g1   = (const float*)d_in[8];
  const float* bt1  = (const float*)d_in[9];
  const float* m1   = (const float*)d_in[10];
  const float* v1   = (const float*)d_in[11];
  const float* W2   = (const float*)d_in[12];
  const float* b2   = (const float*)d_in[13];
  const float* g2   = (const float*)d_in[14];
  const float* bt2  = (const float*)d_in[15];
  const float* m2   = (const float*)d_in[16];
  const float* v2   = (const float*)d_in[17];
  const float* Wr1  = (const float*)d_in[18];
  const float* br1  = (const float*)d_in[19];
  const float* Wr2  = (const float*)d_in[20];
  const float* br2  = (const float*)d_in[21];
  float* out = (float*)d_out;

  // workspace carve
  uint8_t* w = (uint8_t*)d_ws;
  size_t off = 0;
  float*    h    = (float*)(w + off);   off += alignup((size_t)N_NODES * DIM * 4);
  float*    z    = (float*)(w + off);   off += alignup((size_t)N_NODES * DIM * 4);
  uint32_t* zb   = (uint32_t*)(w + off);off += alignup((size_t)N_NODES * DIM * 2);   // bf16 pairs
  uint32_t* y1b  = (uint32_t*)(w + off);off += alignup((size_t)N_NODES * DIM2 * 2);
  uint32_t* WeP  = (uint32_t*)(w + off);off += alignup((size_t)DIM * DIM * 2);
  uint32_t* W1P  = (uint32_t*)(w + off);off += alignup((size_t)NLAYER * DIM * DIM2 * 2);
  uint32_t* W2P  = (uint32_t*)(w + off);off += alignup((size_t)NLAYER * DIM2 * DIM * 2);
  float*    embS = (float*)(w + off);   off += alignup(DIM * 4);
  float*    embH = (float*)(w + off);   off += alignup(DIM * 4);
  float*    s1   = (float*)(w + off);   off += alignup((size_t)NLAYER * DIM2 * 4);
  float*    h1   = (float*)(w + off);   off += alignup((size_t)NLAYER * DIM2 * 4);
  float*    s2   = (float*)(w + off);   off += alignup((size_t)NLAYER * DIM * 4);
  float*    h2   = (float*)(w + off);   off += alignup((size_t)NLAYER * DIM * 4);
  float*    pooled=(float*)(w + off);   off += alignup((size_t)NGRAPH * DIM * 4);

  const int Mtiles = N_NODES / 16;                 // 3125 (exact)

  // 1) fold BN params
  prep_params<<<1, 256, 0, stream>>>(be, b1, g1, bt1, m1, v1, b2, g2, bt2, m2, v2,
                                     embS, embH, s1, h1, s2, h2);
  // 2) pack weights into B-fragment layout
  pack_w<<<(DIM / 16) * (DIM / 32), 32, 0, stream>>>(We, WeP, DIM, DIM);
  for (int l = 0; l < NLAYER; ++l) {
    pack_w<<<(DIM2 / 16) * (DIM / 32), 32, 0, stream>>>(
        W1 + (size_t)l * DIM * DIM2, W1P + (size_t)l * DIM * DIM2 / 2, DIM, DIM2);
    pack_w<<<(DIM / 16) * (DIM2 / 32), 32, 0, stream>>>(
        W2 + (size_t)l * DIM2 * DIM, W2P + (size_t)l * DIM2 * DIM / 2, DIM2, DIM);
  }
  // 3) embed: h = x @ We + be   (x -> bf16 reuses zb)
  {
    int ndw = N_NODES * DIM / 2;
    f32_to_bf16<<<(ndw + 255) / 256, 256, 0, stream>>>((const float2*)x, zb, ndw);
    gemm_wmma<false, false><<<Mtiles, 256, 0, stream>>>(zb, WeP, embS, embH, h, DIM, DIM);
  }
  // 4) GIN layers
  for (int l = 0; l < NLAYER; ++l) {
    int n4 = N_NODES * DIM / 4;
    copy_f4<<<(n4 + 255) / 256, 256, 0, stream>>>((const float4*)h, (float4*)z, n4);
    scatter_add<<<(N_EDGES * 32 + 255) / 256, 256, 0, stream>>>(h, c2, u2, z);
    int ndw = N_NODES * DIM / 2;
    f32_to_bf16<<<(ndw + 255) / 256, 256, 0, stream>>>((const float2*)z, zb, ndw);
    // y1 = relu(bn1(z @ W1 + b1))  -> bf16
    gemm_wmma<true, true><<<Mtiles, 256, 0, stream>>>(
        zb, W1P + (size_t)l * DIM * DIM2 / 2, s1 + l * DIM2, h1 + l * DIM2,
        y1b, DIM, DIM2);
    // h = relu(bn2(y1 @ W2 + b2))  -> fp32
    gemm_wmma<true, false><<<Mtiles, 256, 0, stream>>>(
        y1b, W2P + (size_t)l * DIM2 * DIM / 2, s2 + l * DIM, h2 + l * DIM,
        h, DIM2, DIM);
  }
  // 5) pooling + readout
  zero_buf<<<(NGRAPH * DIM + 255) / 256, 256, 0, stream>>>(pooled, NGRAPH * DIM);
  pool_add<<<(N_NODES * 32 + 255) / 256, 256, 0, stream>>>(h, batch, pooled);
  readout<<<NGRAPH, 128, 0, stream>>>(pooled, Wr1, br1, Wr2, br2, out);
}
